// BidirectionalMambaEncoder_25022479466698
// MI455X (gfx1250) — matrix operand order repaired and checked by reading, not compiled, not run on hardware
//
#include <hip/hip_runtime.h>
#include <hip/hip_bf16.h>

#define D_MODEL   512
#define N_LAYERS  4
#define D_STATE   16
#define D_CONV    4
#define HEADDIM   64
#define D_INNER   1024
#define NHEADS    16
#define CONV_DIM  1056
#define D_IN_PROJ 2096
#define ZX_LD     2112            // D_IN_PROJ padded to a multiple of 32 for 32x32 GEMM tiles
#define D_INTER   256
#define BATCH     4
#define SEQ       2048
#define MROWS     (BATCH * SEQ)   // 8192 token rows
#define CHUNK     64
#define NCHUNK    (SEQ / CHUNK)   // 32 chunks -> sequential depth 32 instead of 2048

typedef _Float16 half8_t  __attribute__((ext_vector_type(8)));
typedef _Float16 half16_t __attribute__((ext_vector_type(16)));
typedef float    float8_t __attribute__((ext_vector_type(8)));

__device__ __forceinline__ float sigmoidf_(float x) { return 1.0f / (1.0f + __expf(-x)); }
__device__ __forceinline__ float siluf_(float x)    { return x * sigmoidf_(x); }

// ---------------------------------------------------------------------------
// Elementwise helpers
// ---------------------------------------------------------------------------
__global__ void cvt_f32_to_f16_kernel(const float* __restrict__ in, _Float16* __restrict__ out, long n) {
  long i = (long)blockIdx.x * blockDim.x + threadIdx.x;
  if (i < n) out[i] = (_Float16)in[i];
}

__global__ void copy_kernel(float* __restrict__ dst, const float* __restrict__ src, long n) {
  long i = (long)blockIdx.x * blockDim.x + threadIdx.x;
  if (i < n) dst[i] = src[i];
}

__global__ void add_inplace_kernel(float* __restrict__ dst, const float* __restrict__ src, long n) {
  long i = (long)blockIdx.x * blockDim.x + threadIdx.x;
  if (i < n) dst[i] += src[i];
}

// ---------------------------------------------------------------------------
// LayerNorm over D_MODEL=512, one wave32 per row, 16 elems/lane.
// f32-out variant (final LN) and f16-out variant (feeds WMMA GEMM directly).
// ---------------------------------------------------------------------------
__global__ void layernorm_kernel(const float* __restrict__ in, const float* __restrict__ w,
                                 const float* __restrict__ b, float* __restrict__ out, int rows) {
  const int wave = threadIdx.x >> 5;
  const int lane = threadIdx.x & 31;
  long row = (long)blockIdx.x * 8 + wave;
  if (row >= rows) return;
  const float* r = in + row * (long)D_MODEL;
  float v[16];
  float s = 0.f, s2 = 0.f;
#pragma unroll
  for (int i = 0; i < 16; ++i) { v[i] = r[lane + i * 32]; s += v[i]; s2 += v[i] * v[i]; }
#pragma unroll
  for (int m = 16; m >= 1; m >>= 1) { s += __shfl_xor(s, m, 32); s2 += __shfl_xor(s2, m, 32); }
  float mu  = s * (1.0f / D_MODEL);
  float var = s2 * (1.0f / D_MODEL) - mu * mu;
  float inv = rsqrtf(var + 1e-5f);
  float* o = out + row * (long)D_MODEL;
#pragma unroll
  for (int i = 0; i < 16; ++i) { int c = lane + i * 32; o[c] = (v[i] - mu) * inv * w[c] + b[c]; }
}

__global__ void layernorm_f16_kernel(const float* __restrict__ in, const float* __restrict__ w,
                                     const float* __restrict__ b, _Float16* __restrict__ out, int rows) {
  const int wave = threadIdx.x >> 5;
  const int lane = threadIdx.x & 31;
  long row = (long)blockIdx.x * 8 + wave;
  if (row >= rows) return;
  const float* r = in + row * (long)D_MODEL;
  float v[16];
  float s = 0.f, s2 = 0.f;
#pragma unroll
  for (int i = 0; i < 16; ++i) { v[i] = r[lane + i * 32]; s += v[i]; s2 += v[i] * v[i]; }
#pragma unroll
  for (int m = 16; m >= 1; m >>= 1) { s += __shfl_xor(s, m, 32); s2 += __shfl_xor(s2, m, 32); }
  float mu  = s * (1.0f / D_MODEL);
  float var = s2 * (1.0f / D_MODEL) - mu * mu;
  float inv = rsqrtf(var + 1e-5f);
  _Float16* o = out + row * (long)D_MODEL;
#pragma unroll
  for (int i = 0; i < 16; ++i) { int c = lane + i * 32; o[c] = (_Float16)((v[i] - mu) * inv * w[c] + b[c]); }
}

// ---------------------------------------------------------------------------
// WMMA GEMM: C(M, 32*N32) = A(M,K) @ W(32*N32, K)^T.
// One wave computes a 32x32 register tile: 2 A-frags x 2 B-frags, 4 WMMAs/k-step.
// Wave-uniform early exit keeps EXEC all-ones at every WMMA.
// ---------------------------------------------------------------------------
__global__ void gemm_wmma_f16_2x2_kernel(const _Float16* __restrict__ A,
                                         const _Float16* __restrict__ W,
                                         float* __restrict__ C,
                                         int Mr, int N32, int K, int ldC) {
  const int lane = threadIdx.x & 31;
  const int wave = threadIdx.x >> 5;
  const long tile  = (long)blockIdx.x * 8 + wave;
  const long total = (long)(Mr >> 5) * N32;
  if (tile >= total) return;                 // wave-uniform
  const int tm = (int)(tile / N32);
  const int tn = (int)(tile % N32);
  const int m0 = tm << 5, n0 = tn << 5;
  const int r = lane & 15;
  const int g = lane >> 4;

  const _Float16* Arow0 = A + (long)(m0 + r) * K;
  const _Float16* Arow1 = A + (long)(m0 + 16 + r) * K;
  const _Float16* Wrow0 = W + (long)(n0 + r) * K;
  const _Float16* Wrow1 = W + (long)(n0 + 16 + r) * K;

  float8_t acc00 = {}, acc01 = {}, acc10 = {}, acc11 = {};
  for (int k0 = 0; k0 < K; k0 += 32) {
    __builtin_prefetch(Arow0 + k0 + 128, 0, 0);
    __builtin_prefetch(Wrow0 + k0 + 128, 0, 0);
    // A fragment: contiguous K-run of 16 per lane group: K = 16g + j
    half8_t p00 = *(const half8_t*)(Arow0 + k0 + 16 * g);
    half8_t p01 = *(const half8_t*)(Arow0 + k0 + 16 * g + 8);
    half8_t p10 = *(const half8_t*)(Arow1 + k0 + 16 * g);
    half8_t p11 = *(const half8_t*)(Arow1 + k0 + 16 * g + 8);
    // B fragment: contiguous K-run of 16 per lane group: K = 16g + j
    half16_t b0 = *(const half16_t*)(Wrow0 + k0 + 16 * g);
    half16_t b1 = *(const half16_t*)(Wrow1 + k0 + 16 * g);
    half16_t a0, a1;
#pragma unroll
    for (int j = 0; j < 8; ++j) { a0[j] = p00[j]; a0[j + 8] = p01[j]; a1[j] = p10[j]; a1[j + 8] = p11[j]; }
    acc00 = __builtin_amdgcn_wmma_f32_16x16x32_f16(false, a0, false, b0, (short)0, acc00, false, false);
    acc01 = __builtin_amdgcn_wmma_f32_16x16x32_f16(false, a0, false, b1, (short)0, acc01, false, false);
    acc10 = __builtin_amdgcn_wmma_f32_16x16x32_f16(false, a1, false, b0, (short)0, acc10, false, false);
    acc11 = __builtin_amdgcn_wmma_f32_16x16x32_f16(false, a1, false, b1, (short)0, acc11, false, false);
  }
  // D layout: lane N = lane&15 ; VGPR i -> M = i + 8*(lane>=16)
  const int mb0 = m0 + (g << 3);
  const int mb1 = m0 + 16 + (g << 3);
#pragma unroll
  for (int i = 0; i < 8; ++i) {
    C[(long)(mb0 + i) * ldC + n0 + r]      = acc00[i];
    C[(long)(mb0 + i) * ldC + n0 + 16 + r] = acc01[i];
    C[(long)(mb1 + i) * ldC + n0 + r]      = acc10[i];
    C[(long)(mb1 + i) * ldC + n0 + 16 + r] = acc11[i];
  }
}

// ---------------------------------------------------------------------------
// dt = softplus(zx[...,2080+h] + dtb[h]); dA = exp(dt * (-exp(alog[h])))
// ---------------------------------------------------------------------------
__global__ void dt_softplus_kernel(const float* __restrict__ zx, const float* __restrict__ dtb,
                                   const float* __restrict__ alog,
                                   float* __restrict__ dt, float* __restrict__ dA, long rows) {
  long i = (long)blockIdx.x * blockDim.x + threadIdx.x;
  if (i >= rows * NHEADS) return;
  int  h   = (int)(i & (NHEADS - 1));
  long row = i >> 4;
  float raw = zx[row * (long)ZX_LD + (D_INNER + CONV_DIM) + h] + dtb[h];
  float sp  = (raw > 20.0f) ? raw : log1pf(__expf(raw));
  float A   = -__expf(alog[h]);
  dt[i] = sp;
  dA[i] = __expf(sp * A);
}

// ---------------------------------------------------------------------------
// Depthwise causal conv(4) + bias + SiLU. rev=1 runs anti-causal (time-flipped layer).
// ---------------------------------------------------------------------------
__global__ void conv_silu_kernel(const float* __restrict__ zx, const float* __restrict__ cw,
                                 const float* __restrict__ cb, float* __restrict__ xBC, int rev) {
  long i = (long)blockIdx.x * blockDim.x + threadIdx.x;
  const long total = (long)MROWS * CONV_DIM;
  if (i >= total) return;
  int  c  = (int)(i % CONV_DIM);
  long bt = i / CONV_DIM;
  int  t  = (int)(bt % SEQ);
  long bb = bt / SEQ;
  const float* w = cw + (long)c * D_CONV;
  float acc = 0.f;
  if (!rev) {
#pragma unroll
    for (int k = 0; k < D_CONV; ++k) {
      int tt = t - (D_CONV - 1) + k;
      if (tt >= 0) acc += w[k] * zx[(bb * SEQ + tt) * (long)ZX_LD + D_INNER + c];
    }
  } else {
#pragma unroll
    for (int k = 0; k < D_CONV; ++k) {
      int tt = t + (D_CONV - 1) - k;
      if (tt < SEQ) acc += w[k] * zx[(bb * SEQ + tt) * (long)ZX_LD + D_INNER + c];
    }
  }
  acc += cb[c];
  xBC[i] = siluf_(acc);
}

// ---------------------------------------------------------------------------
// Chunked SSD selective scan.
// Phase A: intra-chunk scan with h_in=0. Per-step operands (B[16], C[16], x[64])
//   are staged into LDS with a 2-deep pipeline of GLOBAL_LOAD_ASYNC_TO_LDS_B128
//   (CDNA5 async DMA, ASYNCcnt domain): the DMA for step s+2 overlaps the
//   compute of step s+1, hiding global latency on the scan's serial chain.
//   Invariant: 2 async loads always in flight -> s_wait_asynccnt 0x1 == "step
//   s landed". Tail issues clamped dummy loads; S_ENDPGM's implicit wait-idle
//   drains them.
// Phase B (64 blocks, 32 sequential steps): h_in[c+1] = P_c * h_in[c] + S_c.
// Phase C (parallel): y_t += cum_t * (C_t . h_in[chunk(t)]).
// ---------------------------------------------------------------------------
__global__ void scan_chunk_local_kernel(const float* __restrict__ xBC, const float* __restrict__ dt,
                                        const float* __restrict__ dA, const float* __restrict__ Dp,
                                        float* __restrict__ y, float* __restrict__ Sbuf,
                                        float* __restrict__ cumbuf, int rev) {
  const int c  = blockIdx.x & (NCHUNK - 1);
  const int bh = blockIdx.x >> 5;          // NCHUNK == 32
  const int b = bh >> 4, h = bh & 15;
  const int p = threadIdx.x;               // 0..63
  // sbuf[buf][0..15]=B, [16..31]=C, [32..95]=x  (96 floats = 24 lanes x 16B)
  __shared__ float sbuf[2][96];
  const unsigned lds_base = (unsigned)(unsigned long long)(uintptr_t)&sbuf[0][0];

  auto issue = [&](int stp, int bufI) {
    if (threadIdx.x < 24) {                // wave-0 lanes only issue the DMA
      const int sg2 = c * CHUNK + stp;
      const int t2  = rev ? (SEQ - 1 - sg2) : sg2;
      const long base2 = (long)b * SEQ + t2;
      const float* gsrc = (threadIdx.x < 8)
          ? (xBC + base2 * CONV_DIM + D_INNER + threadIdx.x * 4)            // B,C (32 floats)
          : (xBC + base2 * CONV_DIM + h * HEADDIM + (threadIdx.x - 8) * 4); // x (64 floats)
      const unsigned lds = lds_base + (unsigned)(bufI * 96 * 4 + threadIdx.x * 16);
      asm volatile("global_load_async_to_lds_b128 %0, %1, off"
                   :: "v"(lds), "v"(gsrc) : "memory");
    }
  };

  float st[D_STATE];
#pragma unroll
  for (int n = 0; n < D_STATE; ++n) st[n] = 0.f;
  float cum = 1.f;
  const float Dh = Dp[h];

  issue(0, 0);
  issue(CHUNK > 1 ? 1 : 0, 1);
  for (int step = 0; step < CHUNK; ++step) {
    const int buf = step & 1;
    const int sg = c * CHUNK + step;
    const int t = rev ? (SEQ - 1 - sg) : sg;
    const long base = (long)b * SEQ + t;
    // wait until only the (step+1) DMA is outstanding -> step's buffer landed
    asm volatile("s_wait_asynccnt 0x1" ::: "memory");
    __syncthreads();
    const float x   = sbuf[buf][32 + p];
    const float dtv = dt[base * NHEADS + h];
    const float dav = dA[base * NHEADS + h];
    cum *= dav;
    const float dx = dtv * x;
    float acc = 0.f;
#pragma unroll
    for (int n = 0; n < D_STATE; ++n) {
      st[n] = st[n] * dav + dx * sbuf[buf][n];
      acc += st[n] * sbuf[buf][D_STATE + n];
    }
    y[base * D_INNER + h * HEADDIM + p] = acc + Dh * x;
    if (p == 0) cumbuf[(long)bh * SEQ + sg] = cum;
    __syncthreads();                        // all readers of `buf` done before overwrite
    int nxt = step + 2; if (nxt > CHUNK - 1) nxt = CHUNK - 1;
    issue(nxt, buf);                        // keeps 2 loads in flight (dummy at tail)
  }
  float* S = Sbuf + ((long)bh * NCHUNK + c) * (HEADDIM * D_STATE) + p * D_STATE;
#pragma unroll
  for (int n = 0; n < D_STATE; ++n) S[n] = st[n];
}

__global__ void scan_chunk_prefix_kernel(const float* __restrict__ Sbuf,
                                         const float* __restrict__ cumbuf,
                                         float* __restrict__ hinbuf) {
  const int bh = blockIdx.x;               // B*NHEADS = 64
  const int p  = threadIdx.x;              // 0..63
  float hin[D_STATE];
#pragma unroll
  for (int n = 0; n < D_STATE; ++n) hin[n] = 0.f;
  for (int c = 0; c < NCHUNK; ++c) {
    float* H = hinbuf + ((long)bh * NCHUNK + c) * (HEADDIM * D_STATE) + p * D_STATE;
#pragma unroll
    for (int n = 0; n < D_STATE; ++n) H[n] = hin[n];
    const float P = cumbuf[(long)bh * SEQ + c * CHUNK + (CHUNK - 1)];  // chunk total decay
    const float* S = Sbuf + ((long)bh * NCHUNK + c) * (HEADDIM * D_STATE) + p * D_STATE;
#pragma unroll
    for (int n = 0; n < D_STATE; ++n) hin[n] = P * hin[n] + S[n];
  }
}

__global__ void scan_chunk_fix_kernel(const float* __restrict__ xBC, const float* __restrict__ cumbuf,
                                      const float* __restrict__ hinbuf, float* __restrict__ y, int rev) {
  const int c  = blockIdx.x & (NCHUNK - 1);
  const int bh = blockIdx.x >> 5;
  if (c == 0) return;                      // h_in == 0 for the first chunk (block-uniform exit)
  const int b = bh >> 4, h = bh & 15;
  const int p = threadIdx.x;
  __shared__ float sC[D_STATE];
  float hin[D_STATE];
  const float* H = hinbuf + ((long)bh * NCHUNK + c) * (HEADDIM * D_STATE) + p * D_STATE;
#pragma unroll
  for (int n = 0; n < D_STATE; ++n) hin[n] = H[n];
  for (int step = 0; step < CHUNK; ++step) {
    const int sg = c * CHUNK + step;
    const int t = rev ? (SEQ - 1 - sg) : sg;
    const long base = (long)b * SEQ + t;
    if (threadIdx.x < D_STATE)
      sC[threadIdx.x] = xBC[base * CONV_DIM + D_INNER + D_STATE + threadIdx.x];
    __syncthreads();
    const float cum = cumbuf[(long)bh * SEQ + sg];
    float dotv = 0.f;
#pragma unroll
    for (int n = 0; n < D_STATE; ++n) dotv += hin[n] * sC[n];
    y[base * D_INNER + h * HEADDIM + p] += cum * dotv;
    __syncthreads();
  }
}

// ---------------------------------------------------------------------------
// y = rmsnorm(y * silu(z)) * gw over D_INNER; writes f16 GEMM operand directly.
// ---------------------------------------------------------------------------
__global__ void gate_rmsnorm_f16_kernel(float* __restrict__ y, const float* __restrict__ zx,
                                        const float* __restrict__ gw, _Float16* __restrict__ out,
                                        int rows) {
  const int wave = threadIdx.x >> 5;
  const int lane = threadIdx.x & 31;
  long row = (long)blockIdx.x * 8 + wave;
  if (row >= rows) return;
  float* yr = y + row * (long)D_INNER;
  const float* zr = zx + row * (long)ZX_LD;
  float s2 = 0.f;
  for (int i = lane; i < D_INNER; i += 32) {
    float z = zr[i];
    float v = yr[i] * siluf_(z);
    yr[i] = v;
    s2 += v * v;
  }
#pragma unroll
  for (int m = 16; m >= 1; m >>= 1) s2 += __shfl_xor(s2, m, 32);
  float scale = rsqrtf(s2 * (1.0f / D_INNER) + 1e-5f);
  _Float16* o = out + row * (long)D_INNER;
  for (int i = lane; i < D_INNER; i += 32) o[i] = (_Float16)(yr[i] * scale * gw[i]);
}

// ---------------------------------------------------------------------------
// SwiGLU -> f16 GEMM operand: out[row, j] = hh[row, j] * silu(hh[row, 256 + j])
// ---------------------------------------------------------------------------
__global__ void glu_f16_kernel(const float* __restrict__ hh, _Float16* __restrict__ out, long rows) {
  long i = (long)blockIdx.x * blockDim.x + threadIdx.x;
  if (i >= rows * D_INTER) return;
  long row = i / D_INTER;
  int  j   = (int)(i % D_INTER);
  float a = hh[row * (2 * D_INTER) + j];
  float g = hh[row * (2 * D_INTER) + D_INTER + j];
  out[row * (long)D_INTER + j] = (_Float16)(a * siluf_(g));
}

// ---------------------------------------------------------------------------
// Host orchestration
// ---------------------------------------------------------------------------
static inline unsigned blocks_for(long n, int bs) { return (unsigned)((n + bs - 1) / bs); }

extern "C" void kernel_launch(void* const* d_in, const int* in_sizes, int n_in,
                              void* d_out, int out_size, void* d_ws, size_t ws_size,
                              hipStream_t stream) {
  (void)in_sizes; (void)n_in; (void)out_size; (void)ws_size;

  const float* x_in = (const float*)d_in[0];
  const float* n1w  = (const float*)d_in[1];
  const float* n1b  = (const float*)d_in[2];
  const float* inw  = (const float*)d_in[3];
  const float* cw   = (const float*)d_in[4];
  const float* cb   = (const float*)d_in[5];
  const float* dtb  = (const float*)d_in[6];
  const float* alog = (const float*)d_in[7];
  const float* Dp   = (const float*)d_in[8];
  const float* gw   = (const float*)d_in[9];
  const float* ow   = (const float*)d_in[10];
  const float* n2w  = (const float*)d_in[11];
  const float* n2b  = (const float*)d_in[12];
  const float* f1w  = (const float*)d_in[13];
  const float* f2w  = (const float*)d_in[14];
  const float* nfw  = (const float*)d_in[15];
  const float* nfb  = (const float*)d_in[16];
  float* out = (float*)d_out;

  const long M = MROWS;

  size_t off = 0;
  auto alloc_f = [&](size_t elems) -> float* {
    float* p = (float*)((char*)d_ws + off);
    off = (off + elems * sizeof(float) + 255) & ~(size_t)255;
    return p;
  };
  auto alloc_h = [&](size_t elems) -> _Float16* {
    _Float16* p = (_Float16*)((char*)d_ws + off);
    off = (off + elems * sizeof(_Float16) + 255) & ~(size_t)255;
    return p;
  };

  float* resid  = alloc_f((size_t)M * D_MODEL);
  float* xbuf   = alloc_f((size_t)M * D_MODEL);
  float* hbuf   = alloc_f((size_t)M * D_MODEL);       // proj / ffn-hidden scratch
  float* zx     = alloc_f((size_t)M * ZX_LD);         // padded in_proj output
  float* xBC    = alloc_f((size_t)M * CONV_DIM);
  float* dtbuf  = alloc_f((size_t)M * NHEADS);
  float* dabuf  = alloc_f((size_t)M * NHEADS);
  float* ybuf   = alloc_f((size_t)M * D_INNER);
  float* Sbuf   = alloc_f((size_t)BATCH * NHEADS * NCHUNK * HEADDIM * D_STATE);
  float* hinbuf = alloc_f((size_t)BATCH * NHEADS * NCHUNK * HEADDIM * D_STATE);
  float* cumbuf = alloc_f((size_t)BATCH * NHEADS * SEQ);
  _Float16* a16   = alloc_h((size_t)M * D_INNER);     // GEMM A operand (max M x 1024)
  _Float16* inw16 = alloc_h((size_t)N_LAYERS * ZX_LD * D_MODEL);  // padded rows 2096..2111: garbage; their GEMM outputs are never read
  _Float16* ow16  = alloc_h((size_t)N_LAYERS * D_MODEL * D_INNER);
  _Float16* f1w16 = alloc_h((size_t)N_LAYERS * 2 * D_INTER * D_MODEL);
  _Float16* f2w16 = alloc_h((size_t)N_LAYERS * D_MODEL * D_INTER);

  auto cvt = [&](const float* src, _Float16* dst, long n) {
    cvt_f32_to_f16_kernel<<<blocks_for(n, 256), 256, 0, stream>>>(src, dst, n);
  };
  auto gemm = [&](const _Float16* A, const _Float16* W, float* C, int Mr, int N32, int K, int ldC) {
    long tiles = (long)(Mr / 32) * N32;
    gemm_wmma_f16_2x2_kernel<<<blocks_for(tiles, 8), 256, 0, stream>>>(A, W, C, Mr, N32, K, ldC);
  };

  // weights -> fp16 (recomputed every call: deterministic). in_proj converted
  // per layer into padded (ZX_LD x D_MODEL) slots.
  for (int l = 0; l < N_LAYERS; ++l)
    cvt(inw + (size_t)l * D_IN_PROJ * D_MODEL, inw16 + (size_t)l * ZX_LD * D_MODEL,
        (long)D_IN_PROJ * D_MODEL);
  cvt(ow,  ow16,  (long)N_LAYERS * D_MODEL * D_INNER);
  cvt(f1w, f1w16, (long)N_LAYERS * 2 * D_INTER * D_MODEL);
  cvt(f2w, f2w16, (long)N_LAYERS * D_MODEL * D_INTER);

  // residual = x (layer 0)
  copy_kernel<<<blocks_for(M * D_MODEL, 256), 256, 0, stream>>>(resid, x_in, M * D_MODEL);

  for (int l = 0; l < N_LAYERS; ++l) {
    const int rev = l & 1;  // odd layers: reversed conv + scan; all other ops are pointwise in t
    if (l > 0)
      add_inplace_kernel<<<blocks_for(M * D_MODEL, 256), 256, 0, stream>>>(resid, xbuf, M * D_MODEL);

    // zxbcdt = LN1(residual) @ inw^T   (LN fused to f16 operand)
    layernorm_f16_kernel<<<blocks_for(M, 8), 256, 0, stream>>>(resid, n1w + l * D_MODEL, n1b + l * D_MODEL, a16, (int)M);
    gemm(a16, inw16 + (size_t)l * ZX_LD * D_MODEL, zx, (int)M, ZX_LD / 32, D_MODEL, ZX_LD);

    // dt / dA
    dt_softplus_kernel<<<blocks_for(M * NHEADS, 256), 256, 0, stream>>>(
        zx, dtb + l * NHEADS, alog + l * NHEADS, dtbuf, dabuf, M);

    // depthwise conv + SiLU
    conv_silu_kernel<<<blocks_for(M * (long)CONV_DIM, 256), 256, 0, stream>>>(
        zx, cw + (size_t)l * CONV_DIM * D_CONV, cb + l * CONV_DIM, xBC, rev);

    // chunked selective scan: local (async-LDS pipelined) -> prefix -> correction
    scan_chunk_local_kernel<<<BATCH * NHEADS * NCHUNK, HEADDIM, 0, stream>>>(
        xBC, dtbuf, dabuf, Dp + l * NHEADS, ybuf, Sbuf, cumbuf, rev);
    scan_chunk_prefix_kernel<<<BATCH * NHEADS, HEADDIM, 0, stream>>>(Sbuf, cumbuf, hinbuf);
    scan_chunk_fix_kernel<<<BATCH * NHEADS * NCHUNK, HEADDIM, 0, stream>>>(
        xBC, cumbuf, hinbuf, ybuf, rev);

    // residual += rmsnorm(y * silu(z)) @ ow^T
    gate_rmsnorm_f16_kernel<<<blocks_for(M, 8), 256, 0, stream>>>(ybuf, zx, gw + l * D_INNER, a16, (int)M);
    gemm(a16, ow16 + (size_t)l * D_MODEL * D_INNER, hbuf, (int)M, D_MODEL / 32, D_INNER, D_MODEL);
    add_inplace_kernel<<<blocks_for(M * D_MODEL, 256), 256, 0, stream>>>(resid, hbuf, M * D_MODEL);

    // FFN: x = swiglu(LN2(residual) @ f1w^T) @ f2w^T
    layernorm_f16_kernel<<<blocks_for(M, 8), 256, 0, stream>>>(resid, n2w + l * D_MODEL, n2b + l * D_MODEL, a16, (int)M);
    gemm(a16, f1w16 + (size_t)l * 2 * D_INTER * D_MODEL, hbuf, (int)M, (2 * D_INTER) / 32, D_MODEL, 2 * D_INTER);
    glu_f16_kernel<<<blocks_for(M * (long)D_INTER, 256), 256, 0, stream>>>(hbuf, a16, M);
    gemm(a16, f2w16 + (size_t)l * D_MODEL * D_INTER, xbuf, (int)M, D_MODEL / 32, D_INTER, D_MODEL);
  }

  // final residual add + LN -> out
  add_inplace_kernel<<<blocks_for(M * D_MODEL, 256), 256, 0, stream>>>(resid, xbuf, M * D_MODEL);
  layernorm_kernel<<<blocks_for(M, 8), 256, 0, stream>>>(resid, nfw, nfb, out, (int)M);
}
